// Decoder_31679678775681
// MI455X (gfx1250) — compile-verified, matrix operand order checked by reading, not run on hardware
//
#include <hip/hip_runtime.h>
#include <hip/hip_bf16.h>
#include <math.h>

// Problem dims (fixed by the reference)
#define B_   128
#define L_   400
#define V_   50000
#define E_   512
#define H2_  1024
#define M_   1024
#define OOV_ 50
#define VEXT_ (V_ + OOV_)

// LDS strides padded to kill bank conflicts while keeping 16B alignment:
// BSTRIDE*2 = 272 B (17x16), ASTRIDE*2 = 2064 B (129x16); both stride 4 mod 64 dwords.
#define BSTRIDE 136
#define ASTRIDE 1032

typedef __attribute__((ext_vector_type(16))) __bf16 v16bf;
typedef __attribute__((ext_vector_type(8)))  float  v8f;

union BFragQ { v16bf v; uint4 q[2]; };

// float -> bf16 (round to nearest even; data is well-behaved, no NaN path needed)
__device__ __forceinline__ unsigned short f2bf(float f) {
  unsigned int u = __float_as_uint(f);
  u += 0x7FFFu + ((u >> 16) & 1u);
  return (unsigned short)(u >> 16);
}

// ---------------- f32 -> bf16 row-major copy ----------------
__global__ void k_cvt_bf16(const float* __restrict__ X, unsigned short* __restrict__ Xb, int n4) {
  int i = blockIdx.x * 256 + threadIdx.x;
  if (i < n4) {
    const float4 f = ((const float4*)X)[i];
    union { unsigned short s[4]; uint2 u; } o;
    o.s[0] = f2bf(f.x); o.s[1] = f2bf(f.y); o.s[2] = f2bf(f.z); o.s[3] = f2bf(f.w);
    ((uint2*)Xb)[i] = o.u;
  }
}

// ---------------- W_mem -> column-major bf16 copy (tiled transpose) ----------------
__global__ void k_wmem_cvt(const float* __restrict__ W, unsigned short* __restrict__ Wb) {
  __shared__ float tile[16][17];
  int tx = threadIdx.x, ty = threadIdx.y;
  int k0 = blockIdx.x * 16, c0 = blockIdx.y * 16;
  tile[ty][tx] = W[(size_t)(k0 + ty) * H2_ + c0 + tx];
  __syncthreads();
  Wb[(size_t)(c0 + ty) * M_ + k0 + tx] = f2bf(tile[tx][ty]);
}

// ---------------- embedding gather ----------------
__global__ void k_embed(const int* __restrict__ y, const float* __restrict__ emb,
                        float* __restrict__ yemb) {
  int b = blockIdx.x;
  int r = y[b];
  for (int e = threadIdx.x; e < E_; e += blockDim.x)
    yemb[(size_t)b * E_ + e] = emb[(size_t)r * E_ + e];
}

// ---------------- bf16 WMMA GEMM: C[128,N] = Ab[128,K](bf16) @ W[K,N](f32) + bias ----------
// Block = 256 threads (8 waves); one block per 16-wide N stripe covering all 128 rows.
// Double-buffered 128x16 B panels in LDS; staging is split load-burst / store-burst
// so the 8 staging loads clause under a single wait; 4 WMMAs per barrier.
__global__ void k_gemm_bf16(const unsigned short* __restrict__ Ab, const float* __restrict__ Bm,
                            const float* __restrict__ bias, float* __restrict__ C,
                            int N, int K) {
  __shared__ __align__(16) unsigned short Bs[2][16 * BSTRIDE];
  const int t    = threadIdx.x;
  const int wave = t >> 5;
  const int lane = t & 31;
  const int half = lane >> 4;
  const int lr   = lane & 15;
  const int koff = half * 8;
  const int n0   = blockIdx.x * 16;
  const unsigned short* arow = Ab + (size_t)(wave * 16 + lr) * K;

  // staging role of this thread: source row skr (0..15), column scol (0..15)
  const int skr = t >> 4, scol = t & 15;
  const float* bsrc = Bm + (size_t)skr * N + n0 + scol;
  const int sdst = scol * BSTRIDE + skr;

  // stage first panel: burst loads, then converts + LDS stores
  {
    float tmp[8];
#pragma unroll
    for (int r = 0; r < 8; ++r) tmp[r] = bsrc[(size_t)(r * 16) * N];
#pragma unroll
    for (int r = 0; r < 8; ++r) Bs[0][sdst + r * 16] = f2bf(tmp[r]);
  }
  __syncthreads();

  v8f c = {0.f, 0.f, 0.f, 0.f, 0.f, 0.f, 0.f, 0.f};
  for (int kp = 0; kp < K; kp += 128) {
    const int cur = (kp >> 7) & 1;
    if (kp + 128 < K) {
      float tmp[8];
#pragma unroll
      for (int r = 0; r < 8; ++r) tmp[r] = bsrc[(size_t)(kp + 128 + r * 16) * N];
#pragma unroll
      for (int r = 0; r < 8; ++r) Bs[cur ^ 1][sdst + r * 16] = f2bf(tmp[r]);
      if (kp + 256 < K)
        __builtin_prefetch(&bsrc[(size_t)(kp + 256) * N], 0, 1);
    }
    // burst-load 4 K-steps of fragments, then 4 back-to-back WMMAs
    BFragQ aq[4], bq[4];
#pragma unroll
    for (int s = 0; s < 4; ++s) {
      const int ks = s * 32;
      aq[s].q[0] = *(const uint4*)(arow + kp + ks + koff);
      aq[s].q[1] = *(const uint4*)(arow + kp + ks + koff + 16);
      bq[s].q[0] = *(const uint4*)&Bs[cur][lr * BSTRIDE + ks + koff];
      bq[s].q[1] = *(const uint4*)&Bs[cur][lr * BSTRIDE + ks + koff + 16];
    }
#pragma unroll
    for (int s = 0; s < 4; ++s)
      c = __builtin_amdgcn_wmma_f32_16x16x32_bf16(false, aq[s].v, false, bq[s].v, (short)0, c, false, false);
    __syncthreads();
  }
  const int col = n0 + lr;
  const float bb = bias[col];
  const int rbase = wave * 16 + half * 8;                // C layout: VGPR j -> M = j + 8*half
#pragma unroll
  for (int j = 0; j < 8; ++j)
    C[(size_t)(rbase + j) * N + col] = c[j] + bb;
}

// ---------------- GRU gate math (PyTorch r,z,n order) ----------------
__global__ void k_gru(const float* __restrict__ gi, const float* __restrict__ gh,
                      const float* __restrict__ h0, float* __restrict__ h_next,
                      float* __restrict__ h_out) {
  int b = blockIdx.x;
  const float* gib = gi + (size_t)b * 3 * H2_;
  const float* ghb = gh + (size_t)b * 3 * H2_;
  for (int i = threadIdx.x; i < H2_; i += blockDim.x) {
    float r = 1.f / (1.f + expf(-(gib[i] + ghb[i])));
    float z = 1.f / (1.f + expf(-(gib[H2_ + i] + ghb[H2_ + i])));
    float n = tanhf(gib[2 * H2_ + i] + r * ghb[2 * H2_ + i]);
    float hv = (1.f - z) * n + z * h0[(size_t)b * H2_ + i];
    h_next[(size_t)b * H2_ + i] = hv;
    h_out[(size_t)b * H2_ + i]  = hv;
  }
}

// ---------------- fused attention scores (dominant GEMM, ~107 GFLOP) ----------------
// scores[b,l] = v . tanh( memory_bank[b,l,:] @ W_mem + dec_feat[b,:] )
__global__ void k_attn_scores(const float* __restrict__ mb, const unsigned short* __restrict__ Wb,
                              const float* __restrict__ dec_feat, const float* __restrict__ vvec,
                              float* __restrict__ scores) {
  __shared__ __align__(16) unsigned short As[16 * ASTRIDE];
  __shared__ float part[8][16];
  const int t = threadIdx.x;
  const int b = blockIdx.y, l0 = blockIdx.x * 16;

  // stage A tile (16 rows x 1024 cols -> bf16): two bursts of 8 float4 loads,
  // each followed by converts + LDS stores (loads clause under one wait)
  const float* abase = mb + ((size_t)b * L_ + l0) * M_;
#pragma unroll
  for (int g = 0; g < 2; ++g) {
    float4 f[8];
    int base[8];
#pragma unroll
    for (int r = 0; r < 8; ++r) {
      int idx4 = t + (g * 8 + r) * 256;
      int row = idx4 >> 8;                  // M_/4 == 256 float4 per row
      int k4  = idx4 & 255;
      f[r] = *(const float4*)(abase + (size_t)row * M_ + k4 * 4);
      base[r] = row * ASTRIDE + k4 * 4;
    }
#pragma unroll
    for (int r = 0; r < 8; ++r) {
      As[base[r] + 0] = f2bf(f[r].x); As[base[r] + 1] = f2bf(f[r].y);
      As[base[r] + 2] = f2bf(f[r].z); As[base[r] + 3] = f2bf(f[r].w);
    }
  }
  __syncthreads();

  const int wave = t >> 5, lane = t & 31;
  const int half = lane >> 4, lr = lane & 15, koff = half * 8;
  float acc[8];
#pragma unroll
  for (int j = 0; j < 8; ++j) acc[j] = 0.f;

  for (int i = 0; i < 8; ++i) {
    const int nt = (wave * 8 + i) * 16;
    v8f c = {0.f, 0.f, 0.f, 0.f, 0.f, 0.f, 0.f, 0.f};
    const unsigned short* bcol = Wb + (size_t)(nt + lr) * M_;
    for (int kp = 0; kp < M_; kp += 128) {
      // burst-load 4 K-steps (A from LDS, B from the L2-resident bf16 copy)
      BFragQ aq[4], bq[4];
#pragma unroll
      for (int s = 0; s < 4; ++s) {
        const int k = kp + s * 32;
        aq[s].q[0] = *(const uint4*)&As[lr * ASTRIDE + k + koff];
        aq[s].q[1] = *(const uint4*)&As[lr * ASTRIDE + k + koff + 16];
        bq[s].q[0] = *(const uint4*)(bcol + k + koff);
        bq[s].q[1] = *(const uint4*)(bcol + k + koff + 16);
      }
#pragma unroll
      for (int s = 0; s < 4; ++s)
        c = __builtin_amdgcn_wmma_f32_16x16x32_bf16(false, aq[s].v, false, bq[s].v, (short)0, c, false, false);
    }
    const int col = nt + lr;
    const float dv = dec_feat[(size_t)b * H2_ + col];
    const float vv = vvec[col];
#pragma unroll
    for (int j = 0; j < 8; ++j)
      acc[j] += vv * tanhf(c[j] + dv);
  }

  // reduce across the 16 lanes of each half-wave (xor masks < 16 keep halves apart)
#pragma unroll
  for (int j = 0; j < 8; ++j) {
    float a = acc[j];
    for (int off = 1; off < 16; off <<= 1)
      a += __shfl_xor(a, off, 32);
    acc[j] = a;
  }
  if (lr == 0) {
#pragma unroll
    for (int j = 0; j < 8; ++j)
      part[wave][half * 8 + j] = acc[j];
  }
  __syncthreads();
  if (t < 16) {
    float s = 0.f;
#pragma unroll
    for (int w = 0; w < 8; ++w) s += part[w][t];
    scores[(size_t)b * L_ + l0 + t] = s;
  }
}

// ---------------- masked softmax over L, renormalized ----------------
__global__ void k_attn_softmax(const float* __restrict__ scores, const float* __restrict__ mask,
                               float* __restrict__ attn_ws, float* __restrict__ attn_out) {
  __shared__ float red[256];
  int b = blockIdx.x, tid = threadIdx.x;
  const float* srow = scores + (size_t)b * L_;
  const float* mrow = mask + (size_t)b * L_;
  float mx = -1e30f;
  for (int l = tid; l < L_; l += 256) mx = fmaxf(mx, srow[l]);
  red[tid] = mx; __syncthreads();
  for (int s = 128; s > 0; s >>= 1) { if (tid < s) red[tid] = fmaxf(red[tid], red[tid + s]); __syncthreads(); }
  mx = red[0]; __syncthreads();
  float sum = 0.f;
  for (int l = tid; l < L_; l += 256) sum += expf(srow[l] - mx) * mrow[l];
  red[tid] = sum; __syncthreads();
  for (int s = 128; s > 0; s >>= 1) { if (tid < s) red[tid] += red[tid + s]; __syncthreads(); }
  float inv = 1.f / red[0];
  for (int l = tid; l < L_; l += 256) {
    float a = expf(srow[l] - mx) * mrow[l] * inv;
    attn_ws[(size_t)b * L_ + l]  = a;
    attn_out[(size_t)b * L_ + l] = a;
  }
}

// ---------------- context = attn @ memory_bank (bandwidth bound) ----------------
__global__ void k_context(const float* __restrict__ attn, const float* __restrict__ mb,
                          float* __restrict__ ctx_ws, float* __restrict__ ctx_out) {
  int b = blockIdx.x;
  int c0 = threadIdx.x * 4;
  const float* mbb = mb + (size_t)b * L_ * M_;
  const float* arow = attn + (size_t)b * L_;
  float ax = 0.f, ay = 0.f, az = 0.f, aw = 0.f;
  for (int l = 0; l < L_; ++l) {
    float a = arow[l];
    const float4 mv = *(const float4*)(mbb + (size_t)l * M_ + c0);
    ax += a * mv.x; ay += a * mv.y; az += a * mv.z; aw += a * mv.w;
  }
  float4 r = {ax, ay, az, aw};
  *(float4*)(ctx_ws  + (size_t)b * M_ + c0) = r;
  *(float4*)(ctx_out + (size_t)b * M_ + c0) = r;
}

// ---------------- vd_in = [ctx, h_next]; p_gen = sigmoid([ctx,h,yemb].W_pgen + b) ----------------
__global__ void k_concat_pgen(const float* __restrict__ ctx, const float* __restrict__ hn,
                              const float* __restrict__ yemb, const float* __restrict__ Wp,
                              const float* __restrict__ bp, float* __restrict__ vd_in,
                              float* __restrict__ pg_ws, float* __restrict__ pg_out) {
  __shared__ float red[256];
  int b = blockIdx.x, tid = threadIdx.x;
  for (int i = tid; i < M_; i += 256)  vd_in[(size_t)b * (M_ + H2_) + i]      = ctx[(size_t)b * M_ + i];
  for (int i = tid; i < H2_; i += 256) vd_in[(size_t)b * (M_ + H2_) + M_ + i] = hn[(size_t)b * H2_ + i];
  float s = 0.f;
  for (int i = tid; i < M_; i += 256)  s += ctx[(size_t)b * M_ + i] * Wp[i];
  for (int i = tid; i < H2_; i += 256) s += hn[(size_t)b * H2_ + i] * Wp[M_ + i];
  for (int i = tid; i < E_; i += 256)  s += yemb[(size_t)b * E_ + i] * Wp[M_ + H2_ + i];
  red[tid] = s; __syncthreads();
  for (int st = 128; st > 0; st >>= 1) { if (tid < st) red[tid] += red[tid + st]; __syncthreads(); }
  if (tid == 0) {
    float p = 1.f / (1.f + expf(-(red[0] + bp[0])));
    pg_ws[b] = p;
    pg_out[b] = p;
  }
}

// ---------------- softmax over V=50000 (in place) ----------------
__global__ void k_vocab_softmax(float* __restrict__ logits) {
  __shared__ float red[512];
  int b = blockIdx.x, tid = threadIdx.x;
  float* row = logits + (size_t)b * V_;
  float mx = -1e30f;
  for (int i = tid; i < V_; i += 512) mx = fmaxf(mx, row[i]);
  red[tid] = mx; __syncthreads();
  for (int s = 256; s > 0; s >>= 1) { if (tid < s) red[tid] = fmaxf(red[tid], red[tid + s]); __syncthreads(); }
  mx = red[0]; __syncthreads();
  float sum = 0.f;
  for (int i = tid; i < V_; i += 512) sum += expf(row[i] - mx);
  red[tid] = sum; __syncthreads();
  for (int s = 256; s > 0; s >>= 1) { if (tid < s) red[tid] += red[tid + s]; __syncthreads(); }
  float inv = 1.f / red[0];
  for (int i = tid; i < V_; i += 512) row[i] = expf(row[i] - mx) * inv;
}

// ---------------- final distribution + copy-mechanism scatter ----------------
__global__ void k_final(const float* __restrict__ vocab, const float* __restrict__ attn,
                        const float* __restrict__ pg, const int* __restrict__ src_oov,
                        float* __restrict__ out_fd) {
  __shared__ float addv[L_];
  __shared__ int   addi[L_];
  int b = blockIdx.x, tid = threadIdx.x;
  float p = pg[b];
  float* row = out_fd + (size_t)b * VEXT_;
  for (int i = tid; i < VEXT_; i += 256)
    row[i] = (i < V_) ? p * vocab[(size_t)b * V_ + i] : 0.f;
  float q = 1.f - p;
  for (int l = tid; l < L_; l += 256) {
    addv[l] = q * attn[(size_t)b * L_ + l];
    addi[l] = src_oov[(size_t)b * L_ + l];
  }
  __syncthreads();
  if (tid == 0) {
    for (int l = 0; l < L_; ++l)
      row[addi[l]] += addv[l];
  }
}

extern "C" void kernel_launch(void* const* d_in, const int* in_sizes, int n_in,
                              void* d_out, int out_size, void* d_ws, size_t ws_size,
                              hipStream_t stream) {
  (void)in_sizes; (void)n_in; (void)out_size; (void)ws_size;

  const int*   y      = (const int*)d_in[0];
  const float* h      = (const float*)d_in[1];   // [1,B,H2] == [B,H2]
  const float* mb     = (const float*)d_in[2];   // [B,L,M]
  const float* mask   = (const float*)d_in[3];   // [B,L]
  /* d_in[4]: max_num_oovs scalar (compile-time OOV_) */
  const int*   srcoov = (const int*)d_in[5];     // [B,L]
  const float* emb    = (const float*)d_in[6];   // [V,E]
  const float* W_ih   = (const float*)d_in[7];   // [E,3H]
  const float* W_hh   = (const float*)d_in[8];   // [H,3H]
  const float* b_ih   = (const float*)d_in[9];
  const float* b_hh   = (const float*)d_in[10];
  const float* W_mem  = (const float*)d_in[11];  // [M,H]
  const float* W_dec  = (const float*)d_in[12];  // [H,H]
  const float* b_dec  = (const float*)d_in[13];
  const float* vvec   = (const float*)d_in[14];  // [H]
  const float* W_pgen = (const float*)d_in[15];  // [M+H+E,1]
  const float* b_pgen = (const float*)d_in[16];
  const float* W_v1   = (const float*)d_in[17];  // [M+H,H]
  const float* b_v1   = (const float*)d_in[18];
  const float* W_v2   = (const float*)d_in[19];  // [H,V]
  const float* b_v2   = (const float*)d_in[20];

  // workspace layout (float-sized accounting)
  float* wsf    = (float*)d_ws;
  float* yemb   = wsf;  wsf += (size_t)B_ * E_;
  float* gi     = wsf;  wsf += (size_t)B_ * 3 * H2_;
  float* gh     = wsf;  wsf += (size_t)B_ * 3 * H2_;
  float* hnext  = wsf;  wsf += (size_t)B_ * H2_;
  float* decf   = wsf;  wsf += (size_t)B_ * H2_;
  float* scores = wsf;  wsf += (size_t)B_ * L_;
  float* attn   = wsf;  wsf += (size_t)B_ * L_;
  float* ctx    = wsf;  wsf += (size_t)B_ * M_;
  float* vd_in  = wsf;  wsf += (size_t)B_ * (M_ + H2_);
  float* hidden = wsf;  wsf += (size_t)B_ * H2_;
  float* pgen   = wsf;  wsf += (size_t)B_;
  float* logits = wsf;  wsf += (size_t)B_ * V_;
  unsigned short* Wb = (unsigned short*)wsf;     // column-major bf16 W_mem (2 MB)
  wsf += (size_t)(M_ * H2_) / 2;
  unsigned short* Abf = (unsigned short*)wsf;    // reusable bf16 A scratch (512 KB max)
  wsf += (size_t)(B_ * (M_ + H2_)) / 2;

  // output layout (flat concat in reference return order)
  float* out      = (float*)d_out;
  float* out_fd   = out;                               // [B, V+OOV]
  float* out_h    = out_fd  + (size_t)B_ * VEXT_;      // [1,B,H2]
  float* out_ctx  = out_h   + (size_t)B_ * H2_;        // [B,M]
  float* out_attn = out_ctx + (size_t)B_ * M_;         // [B,L]
  float* out_pg   = out_attn + (size_t)B_ * L_;        // [B,1]

  // 0) one-time (per call) W_mem transpose+convert for the attention GEMM
  k_wmem_cvt<<<dim3(M_ / 16, H2_ / 16), dim3(16, 16), 0, stream>>>(W_mem, Wb);
  // 1) embedding gather
  k_embed<<<B_, 256, 0, stream>>>(y, emb, yemb);
  // 2) GRU input/hidden gate GEMMs (bf16 WMMA, f32 accumulate)
  k_cvt_bf16<<<(B_ * E_ / 4 + 255) / 256, 256, 0, stream>>>(yemb, Abf, B_ * E_ / 4);
  k_gemm_bf16<<<3 * H2_ / 16, 256, 0, stream>>>(Abf, W_ih, b_ih, gi, 3 * H2_, E_);
  k_cvt_bf16<<<(B_ * H2_ / 4 + 255) / 256, 256, 0, stream>>>(h, Abf, B_ * H2_ / 4);
  k_gemm_bf16<<<3 * H2_ / 16, 256, 0, stream>>>(Abf, W_hh, b_hh, gh, 3 * H2_, H2_);
  // 3) gate math -> h_next (also streamed to output)
  k_gru<<<B_, 256, 0, stream>>>(gi, gh, h, hnext, out_h);
  // 4) dec_feat = h_next @ W_dec + b_dec
  k_cvt_bf16<<<(B_ * H2_ / 4 + 255) / 256, 256, 0, stream>>>(hnext, Abf, B_ * H2_ / 4);
  k_gemm_bf16<<<H2_ / 16, 256, 0, stream>>>(Abf, W_dec, b_dec, decf, H2_, H2_);
  // 5) fused attention scores
  k_attn_scores<<<dim3(L_ / 16, B_), 256, 0, stream>>>(mb, Wb, decf, vvec, scores);
  // 6) masked, renormalized softmax
  k_attn_softmax<<<B_, 256, 0, stream>>>(scores, mask, attn, out_attn);
  // 7) context (second streaming pass over memory_bank)
  k_context<<<B_, 256, 0, stream>>>(attn, mb, ctx, out_ctx);
  // 8) concat + p_gen
  k_concat_pgen<<<B_, 256, 0, stream>>>(ctx, hnext, yemb, W_pgen, b_pgen, vd_in, pgen, out_pg);
  // 9) hidden = vd_in @ W_v1 + b_v1
  k_cvt_bf16<<<(B_ * (M_ + H2_) / 4 + 255) / 256, 256, 0, stream>>>(vd_in, Abf, B_ * (M_ + H2_) / 4);
  k_gemm_bf16<<<H2_ / 16, 256, 0, stream>>>(Abf, W_v1, b_v1, hidden, H2_, M_ + H2_);
  // 10) logits = hidden @ W_v2 + b_v2 (memory-bound: streams 205 MB coalesced, once)
  k_cvt_bf16<<<(B_ * H2_ / 4 + 255) / 256, 256, 0, stream>>>(hidden, Abf, B_ * H2_ / 4);
  k_gemm_bf16<<<V_ / 16, 256, 0, stream>>>(Abf, W_v2, b_v2, logits, V_, H2_);
  // 11) vocab softmax (in place) + final copy-mechanism scatter
  k_vocab_softmax<<<B_, 512, 0, stream>>>(logits);
  k_final<<<B_, 256, 0, stream>>>(logits, attn, pgen, srcoov, out_fd);
}
